// CrossModalAttention_41094247088222
// MI455X (gfx1250) — matrix operand order repaired and checked by reading, not compiled, not run on hardware
//
#include <hip/hip_runtime.h>
#include <hip/hip_bf16.h>
#include <math.h>

// ---------------------------------------------------------------------------
// CDNA5 (gfx1250) WMMA cross-modal attention pipeline.
// All GEMMs run through v_wmma_f32_16x16x32_f16 (wave32).
// Each wave computes a 32x64 output tile (2x4 WMMA register blocking):
//   per K-step: 6 fragment loads -> 8 WMMAs (21.3 FLOP/byte from cache).
// __launch_bounds__(256, 2) widens the VGPR budget so all fragments stay
// double-buffered (no per-fragment s_wait_loadcnt 0x0 serialization).
// ---------------------------------------------------------------------------

typedef __attribute__((ext_vector_type(16))) _Float16 v16h;
typedef __attribute__((ext_vector_type(8)))  _Float16 v8h;
typedef __attribute__((ext_vector_type(8)))  float    v8f;

#define QD 1024   // QUERY_DIM
#define KD 512    // KEY_DIM
#define HD 1024   // HIDDEN_DIM
#define BB 8
#define LQ 2048
#define LK 2048

// ---------------------------------------------------------------------------
// Fragment loaders per CDNA5 ISA 7.12.2 (16-bit A 16x32, B 32x16):
//   A: lane m=l%16, half=l/16; e<8 -> K=half*8+e ; e>=8 -> K=16+half*8+(e-8)
//   B: lane n=l%16, half=l/16; e<8 -> K=half*8+e ; e>=8 -> K=16+half*8+(e-8)
//   C: vgpr r -> M=r+8*half, N=l%16
// ---------------------------------------------------------------------------
__device__ __forceinline__ v16h load_a_frag(const _Float16* __restrict__ row,
                                            int k0, int half) {
  v8h lo = *(const v8h*)(row + k0 + half * 8);
  v8h hi = *(const v8h*)(row + k0 + 16 + half * 8);
  return __builtin_shufflevector(lo, hi,
           0,1,2,3,4,5,6,7,8,9,10,11,12,13,14,15);
}

__device__ __forceinline__ v16h load_b_frag(const _Float16* __restrict__ row,
                                            int k0, int half) {
  v8h lo = *(const v8h*)(row + k0 + half * 8);
  v8h hi = *(const v8h*)(row + k0 + 16 + half * 8);
  return __builtin_shufflevector(lo, hi,
           0,1,2,3,4,5,6,7,8,9,10,11,12,13,14,15);
}

#define WMMA(cacc, a, b)                                                     \
  cacc = __builtin_amdgcn_wmma_f32_16x16x32_f16(false, (a), false, (b),      \
                                                (short)0, (cacc), false, false)

struct Acc8 { v8f c[2][4]; };

// Wave-level 32x64 tile: C = A(32xK) * Bt(64xK)^T, 2x4 register blocking.
__device__ __forceinline__ Acc8 wmma_tile_32x64(const _Float16* __restrict__ aTile,
                                                long lda,
                                                const _Float16* __restrict__ bTile,
                                                long ldb,
                                                int Kdim, int lane) {
  const int sub  = lane & 15;
  const int half = lane >> 4;
  const _Float16* aR0 = aTile + (long)sub * lda;
  const _Float16* aR1 = aTile + (long)(16 + sub) * lda;
  const _Float16* bR0 = bTile + (long)sub * ldb;
  const _Float16* bR1 = bTile + (long)(16 + sub) * ldb;
  const _Float16* bR2 = bTile + (long)(32 + sub) * ldb;
  const _Float16* bR3 = bTile + (long)(48 + sub) * ldb;
  Acc8 acc = {};
  for (int k0 = 0; k0 < Kdim; k0 += 32) {
    __builtin_prefetch(aR0 + k0 + 512, 0, 1);   // global_prefetch_b8
    __builtin_prefetch(bR0 + k0 + 512, 0, 1);
    v16h a0 = load_a_frag(aR0, k0, half);
    v16h a1 = load_a_frag(aR1, k0, half);
    v16h b0 = load_b_frag(bR0, k0, half);
    v16h b1 = load_b_frag(bR1, k0, half);
    v16h b2 = load_b_frag(bR2, k0, half);
    v16h b3 = load_b_frag(bR3, k0, half);
    WMMA(acc.c[0][0], a0, b0);
    WMMA(acc.c[1][0], a1, b0);
    WMMA(acc.c[0][1], a0, b1);
    WMMA(acc.c[1][1], a1, b1);
    WMMA(acc.c[0][2], a0, b2);
    WMMA(acc.c[1][2], a1, b2);
    WMMA(acc.c[0][3], a0, b3);
    WMMA(acc.c[1][3], a1, b3);
  }
  return acc;
}

// Block = 256 threads = 8 waves arranged 4(M) x 2(N): block tile 128 x 128.
// tileM32 indexes 32-row units; tileN64 indexes 64-col units.
__device__ __forceinline__ void tile_coords(long& tileM32, long& tileN64, int& lane) {
  lane = threadIdx.x & 31;
  const int w = threadIdx.x >> 5;
  tileM32 = blockIdx.y * 4 + (w >> 1);
  tileN64 = blockIdx.x * 2 + (w & 1);
}

// ---------------------------------------------------------------------------
// Element-wise f32 -> f16 convert
// ---------------------------------------------------------------------------
__global__ void cvt_f32_f16(const float* __restrict__ x, _Float16* __restrict__ y,
                            long n) {
  long i = blockIdx.x * (long)blockDim.x + threadIdx.x;
  long stride = gridDim.x * (long)blockDim.x;
  for (; i < n; i += stride) y[i] = (_Float16)x[i];
}

// Transpose-convert weight: W[D,H] f32 -> Wt[H,D] f16 (Wt[h*D+d] = W[d*H+h])
__global__ void cvt_trans_f16(const float* __restrict__ W, _Float16* __restrict__ Wt,
                              int D, int H) {
  long n = (long)D * H;
  long i = blockIdx.x * (long)blockDim.x + threadIdx.x;
  long stride = gridDim.x * (long)blockDim.x;
  for (; i < n; i += stride) {
    long h = i / D, d = i % D;
    Wt[i] = (_Float16)W[d * (long)H + h];
  }
}

// ---------------------------------------------------------------------------
// Generic WMMA GEMM, f16 output: C[M,N] = scale * (A[M,K] @ Bt[N,K]^T) + bias
// grid.z = batch with given element strides.
// ---------------------------------------------------------------------------
__global__ void __launch_bounds__(256, 2)
gemm_wmma_f16(const _Float16* __restrict__ A,
              const _Float16* __restrict__ Bt,
              const float* __restrict__ bias,
              _Float16* __restrict__ C,
              int Kdim, int N, float scale,
              long sA, long sB, long sC) {
  long tileM32, tileN64; int lane;
  tile_coords(tileM32, tileN64, lane);
  const _Float16* a = A + blockIdx.z * sA + tileM32 * 32 * (long)Kdim;
  const _Float16* b = Bt + blockIdx.z * sB + tileN64 * 64 * (long)Kdim;
  Acc8 acc = wmma_tile_32x64(a, Kdim, b, Kdim, Kdim, lane);

  const int sub = lane & 15, half = lane >> 4;
  _Float16* out = C + blockIdx.z * sC;
#pragma unroll
  for (int i = 0; i < 2; ++i) {
#pragma unroll
    for (int j = 0; j < 4; ++j) {
      const long col = tileN64 * 64 + j * 16 + sub;
      const float bval = bias ? bias[col] : 0.0f;
#pragma unroll
      for (int r = 0; r < 8; ++r) {
        long row = tileM32 * 32 + i * 16 + r + 8 * half;
        out[row * (long)N + col] = (_Float16)(acc.c[i][j][r] * scale + bval);
      }
    }
  }
}

// V projection with transposed store: rows are (b*LK + k), out Vt[b][h][k].
__global__ void __launch_bounds__(256, 2)
gemm_wmma_f16_vt(const _Float16* __restrict__ A,
                 const _Float16* __restrict__ Bt,
                 const float* __restrict__ bias,
                 _Float16* __restrict__ Vt,
                 int Kdim, int N /*=HD*/) {
  long tileM32, tileN64; int lane;
  tile_coords(tileM32, tileN64, lane);
  const _Float16* a = A + tileM32 * 32 * (long)Kdim;
  const _Float16* b = Bt + tileN64 * 64 * (long)Kdim;
  Acc8 acc = wmma_tile_32x64(a, Kdim, b, Kdim, Kdim, lane);

  const int sub = lane & 15, half = lane >> 4;
#pragma unroll
  for (int i = 0; i < 2; ++i) {
#pragma unroll
    for (int j = 0; j < 4; ++j) {
      const long col = tileN64 * 64 + j * 16 + sub;      // h
      const float bval = bias[col];
#pragma unroll
      for (int r = 0; r < 8; ++r) {
        long row = tileM32 * 32 + i * 16 + r + 8 * half; // b*LK + k
        long bb  = row / LK;
        long kk  = row - bb * LK;
        Vt[(bb * (long)N + col) * LK + kk] = (_Float16)(acc.c[i][j][r] + bval);
      }
    }
  }
}

// Final GEMM, f32 output with bias + residual: out = A @ Bt^T + bias + resid
__global__ void __launch_bounds__(256, 2)
gemm_wmma_final(const _Float16* __restrict__ A,
                const _Float16* __restrict__ Bt,
                const float* __restrict__ bias,
                const float* __restrict__ resid,
                float* __restrict__ out,
                int Kdim, int N) {
  long tileM32, tileN64; int lane;
  tile_coords(tileM32, tileN64, lane);
  const _Float16* a = A + tileM32 * 32 * (long)Kdim;
  const _Float16* b = Bt + tileN64 * 64 * (long)Kdim;
  Acc8 acc = wmma_tile_32x64(a, Kdim, b, Kdim, Kdim, lane);

  const int sub = lane & 15, half = lane >> 4;
#pragma unroll
  for (int i = 0; i < 2; ++i) {
#pragma unroll
    for (int j = 0; j < 4; ++j) {
      const long col = tileN64 * 64 + j * 16 + sub;
      const float bval = bias[col];
#pragma unroll
      for (int r = 0; r < 8; ++r) {
        long row = tileM32 * 32 + i * 16 + r + 8 * half;
        long idx = row * (long)N + col;
        out[idx] = acc.c[i][j][r] + bval + resid[idx];
      }
    }
  }
}

// ---------------------------------------------------------------------------
// Row softmax over LK f16 scores, in place. One block (256 thr) per row.
// ---------------------------------------------------------------------------
__global__ void softmax_rows(_Float16* __restrict__ S, int ncols) {
  __shared__ float red[256];
  const int tid = threadIdx.x;
  _Float16* p = S + blockIdx.x * (long)ncols;

  float lmax = -3.0e38f;
  for (int i = tid; i < ncols; i += 256) lmax = fmaxf(lmax, (float)p[i]);
  red[tid] = lmax; __syncthreads();
  for (int s = 128; s > 0; s >>= 1) {
    if (tid < s) red[tid] = fmaxf(red[tid], red[tid + s]);
    __syncthreads();
  }
  const float m = red[0];
  __syncthreads();

  float lsum = 0.0f;
  for (int i = tid; i < ncols; i += 256) lsum += __expf((float)p[i] - m);
  red[tid] = lsum; __syncthreads();
  for (int s = 128; s > 0; s >>= 1) {
    if (tid < s) red[tid] += red[tid + s];
    __syncthreads();
  }
  const float inv = 1.0f / red[0];
  __syncthreads();

  for (int i = tid; i < ncols; i += 256)
    p[i] = (_Float16)(__expf((float)p[i] - m) * inv);
}

// ---------------------------------------------------------------------------
// Host launcher
// ---------------------------------------------------------------------------
extern "C" void kernel_launch(void* const* d_in, const int* in_sizes, int n_in,
                              void* d_out, int out_size, void* d_ws, size_t ws_size,
                              hipStream_t stream) {
  (void)in_sizes; (void)n_in; (void)out_size; (void)ws_size;

  const float* query = (const float*)d_in[0];  // [B, LQ, QD]
  const float* key   = (const float*)d_in[1];  // [B, LK, KD]
  const float* Wq    = (const float*)d_in[2];  // [QD, HD]
  const float* bq    = (const float*)d_in[3];
  const float* Wk    = (const float*)d_in[4];  // [KD, HD]
  const float* bk    = (const float*)d_in[5];
  const float* Wv    = (const float*)d_in[6];  // [KD, HD]
  const float* bv    = (const float*)d_in[7];
  const float* Wo    = (const float*)d_in[8];  // [HD, QD]
  const float* bo    = (const float*)d_in[9];
  float* out = (float*)d_out;                  // [B, LQ, QD]

  const long M = (long)BB * LQ;                // 16384 rows for projections
  const float scale = 1.0f / 32.0f;            // HIDDEN_DIM^-0.5

  // Workspace carve-up (f16 elements), 256B aligned
  char* ws = (char*)d_ws;
  size_t off = 0;
  auto carve = [&](size_t elems) {
    _Float16* p = (_Float16*)(ws + off);
    off += (elems * sizeof(_Float16) + 255) & ~(size_t)255;
    return p;
  };
  _Float16* q16  = carve(M * QD);              // query f16
  _Float16* k16  = carve(M * KD);              // key f16
  _Float16* wqt  = carve((size_t)HD * QD);     // Wq^T  [HD, QD]
  _Float16* wkt  = carve((size_t)HD * KD);     // Wk^T  [HD, KD]
  _Float16* wvt  = carve((size_t)HD * KD);     // Wv^T  [HD, KD]
  _Float16* wot  = carve((size_t)QD * HD);     // Wo^T  [QD, HD]
  _Float16* Qp   = carve(M * HD);              // Q projection
  _Float16* Kp   = carve(M * HD);              // K projection
  _Float16* Vt   = carve((size_t)BB * HD * LK);// V projection, transposed per batch
  _Float16* S16  = carve((size_t)BB * LQ * LK);// scores / probabilities
  _Float16* att  = carve(M * HD);              // attended

  dim3 blk(256);

  // 1) converts
  cvt_f32_f16<<<dim3(4096), blk, 0, stream>>>(query, q16, M * QD);
  cvt_f32_f16<<<dim3(2048), blk, 0, stream>>>(key,   k16, M * KD);
  cvt_trans_f16<<<dim3(1024), blk, 0, stream>>>(Wq, wqt, QD, HD);
  cvt_trans_f16<<<dim3(512),  blk, 0, stream>>>(Wk, wkt, KD, HD);
  cvt_trans_f16<<<dim3(512),  blk, 0, stream>>>(Wv, wvt, KD, HD);
  cvt_trans_f16<<<dim3(1024), blk, 0, stream>>>(Wo, wot, HD, QD);

  // Block tile = 128(M) x 128(N)
  // 2) projections: Q = q16 @ Wq + bq ; K = k16 @ Wk + bk ; Vt = (k16 @ Wv + bv)^T
  gemm_wmma_f16<<<dim3(HD / 128, M / 128, 1), blk, 0, stream>>>(
      q16, wqt, bq, Qp, QD, HD, 1.0f, 0, 0, 0);
  gemm_wmma_f16<<<dim3(HD / 128, M / 128, 1), blk, 0, stream>>>(
      k16, wkt, bk, Kp, KD, HD, 1.0f, 0, 0, 0);
  gemm_wmma_f16_vt<<<dim3(HD / 128, M / 128, 1), blk, 0, stream>>>(
      k16, wvt, bv, Vt, KD, HD);

  // 3) scores: S[b] = scale * Q[b] @ K[b]^T   (Bt = Kp rows)
  gemm_wmma_f16<<<dim3(LK / 128, LQ / 128, BB), blk, 0, stream>>>(
      Qp, Kp, nullptr, S16, HD, LK, scale,
      (long)LQ * HD, (long)LK * HD, (long)LQ * LK);

  // 4) softmax over keys
  softmax_rows<<<dim3(BB * LQ), blk, 0, stream>>>(S16, LK);

  // 5) attended: A[b] = P[b] @ V[b]           (Bt = Vt[b] rows, contiguous)
  gemm_wmma_f16<<<dim3(HD / 128, LQ / 128, BB), blk, 0, stream>>>(
      S16, Vt, nullptr, att, LK, HD, 1.0f,
      (long)LQ * LK, (long)HD * LK, (long)LQ * HD);

  // 6) output projection + bias + residual (f32 out)
  gemm_wmma_final<<<dim3(QD / 128, M / 128, 1), blk, 0, stream>>>(
      att, wot, bo, query, out, HD, QD);
}
